// EquivariantUNet_34041910788188
// MI455X (gfx1250) — compile-verified, matrix operand order checked by reading, not run on hardware
//
#include <hip/hip_runtime.h>
#include <hip/hip_bf16.h>

typedef _Float16 h16;
typedef __attribute__((ext_vector_type(16))) _Float16 v16h;
typedef __attribute__((ext_vector_type(8)))  _Float16 v8h;
typedef __attribute__((ext_vector_type(8)))  float    v8f;

// ---------------------------------------------------------------------------
// WMMA fragment loader for V_WMMA_F32_16X16X32_F16 (wave32).
// A-matrix 16x32 f16 layout (ISA 7.12.2): lane L holds row M=L%16.
//   lanes 0-15 : halves j=0..7 -> K=j      ; j=8..15 -> K=j+8  (16..23)
//   lanes 16-31: halves j=0..7 -> K=j+8    ; j=8..15 -> K=j+16 (24..31)
// => two contiguous 8-half (16B) segments at k0+hi and k0+hi+16, hi=8*(L>=16).
// B (32x16) is symmetric with column index across lanes, so the same loader
// works on a column-major (transposed) weight tile: row index = output col.
// ---------------------------------------------------------------------------
__device__ __forceinline__ v16h frag_ld(const h16* p, int k0, int lane) {
  const int hi = (lane & 16) ? 8 : 0;
  const h16* q = p + k0 + hi;
  v8h lo = *(const v8h*)(q);
  v8h hg = *(const v8h*)(q + 16);
  v16h f;
#pragma unroll
  for (int j = 0; j < 8; ++j) { f[j] = lo[j]; f[j + 8] = hg[j]; }
  return f;
}

__device__ __forceinline__ v8f wmma16(v16h a, v16h b, v8f c) {
  return __builtin_amdgcn_wmma_f32_16x16x32_f16(false, a, false, b, (short)0, c,
                                                false, false);
}

// Fast silu: one v_exp_f32 + one v_rcp_f32 (~1e-6 rel err, far below f16
// input quantization of the GEMMs).
__device__ __forceinline__ float silu(float v) {
  return v * __builtin_amdgcn_rcpf(1.f + __expf(-v));
}

// ===========================================================================
// Kernel 1: node MLP   x_t = silu(x@W1+b1)@W2+b2        (N x 64)
// Persistent blocks: 8 waves, each loops over 16-node tiles.
// ===========================================================================
__global__ __launch_bounds__(256) void node_mlp_kernel(
    const float* __restrict__ x, const float* __restrict__ W1,
    const float* __restrict__ B1, const float* __restrict__ W2,
    const float* __restrict__ B2, float* __restrict__ xt, int Nn) {
  __shared__ __align__(16) h16 Wt1[64 * 72];   // [n][k], k=0..63, stride 72
  __shared__ __align__(16) h16 Wt2[64 * 72];
  __shared__ float b1s[64], b2s[64];
  __shared__ __align__(16) h16 At[8][16 * 72];  // per-wave A tile, stride 72

  const int tid = threadIdx.x, w = tid >> 5, lane = tid & 31;
  for (int i = tid; i < 64 * 64; i += 256) {
    int n = i >> 6, k = i & 63;
    Wt1[n * 72 + k] = (h16)W1[k * 64 + n];
    Wt2[n * 72 + k] = (h16)W2[k * 64 + n];
  }
  if (tid < 64) { b1s[tid] = B1[tid]; b2s[tid] = B2[tid]; }
  __syncthreads();  // weights ready; everything below is wave-private

  const int mrow = lane & 15;
  const int half = lane >> 4;          // 0 or 1
  const int hi8  = half << 3;          // +8 rows for upper half-wave
  const int numTiles = (Nn + 15) >> 4;
  const int tileStep = gridDim.x * 8;

  for (int t = blockIdx.x * 8 + w; t < numTiles; t += tileStep) {
    const int n0 = t * 16;
    // --- vectorized A fill: lane (mrow,half) covers row mrow, cols half*32.. ---
    {
      int nn = n0 + mrow;
      v8h* dst = (v8h*)&At[w][mrow * 72 + half * 32];
      if (nn < Nn) {
        const float4* p = (const float4*)(x + (size_t)nn * 64 + half * 32);
#pragma unroll
        for (int jj = 0; jj < 4; ++jj) {
          float4 a0 = p[2 * jj], a1 = p[2 * jj + 1];
          v8h hh;
          hh[0] = (h16)a0.x; hh[1] = (h16)a0.y; hh[2] = (h16)a0.z; hh[3] = (h16)a0.w;
          hh[4] = (h16)a1.x; hh[5] = (h16)a1.y; hh[6] = (h16)a1.z; hh[7] = (h16)a1.w;
          dst[jj] = hh;
        }
      } else {
        v8h z = {};
#pragma unroll
        for (int jj = 0; jj < 4; ++jj) dst[jj] = z;
      }
    }
    v8f acc[4] = {v8f{}, v8f{}, v8f{}, v8f{}};
#pragma unroll
    for (int ks = 0; ks < 2; ++ks) {
      v16h af = frag_ld(&At[w][mrow * 72], ks * 32, lane);
#pragma unroll
      for (int ct = 0; ct < 4; ++ct) {
        v16h bf = frag_ld(&Wt1[(ct * 16 + mrow) * 72], ks * 32, lane);
        acc[ct] = wmma16(af, bf, acc[ct]);
      }
    }
#pragma unroll
    for (int ct = 0; ct < 4; ++ct) {
      int n = ct * 16 + mrow;
      float bias = b1s[n];
#pragma unroll
      for (int r = 0; r < 8; ++r) {
        int M = r + hi8;
        At[w][M * 72 + n] = (h16)silu(acc[ct][r] + bias);
      }
    }
    v8f acc2[4] = {v8f{}, v8f{}, v8f{}, v8f{}};
#pragma unroll
    for (int ks = 0; ks < 2; ++ks) {
      v16h af = frag_ld(&At[w][mrow * 72], ks * 32, lane);
#pragma unroll
      for (int ct = 0; ct < 4; ++ct) {
        v16h bf = frag_ld(&Wt2[(ct * 16 + mrow) * 72], ks * 32, lane);
        acc2[ct] = wmma16(af, bf, acc2[ct]);
      }
    }
#pragma unroll
    for (int r = 0; r < 8; ++r) {
      int M = r + hi8, nn = n0 + M;
      if (nn < Nn) {
#pragma unroll
        for (int ct = 0; ct < 4; ++ct) {
          int n = ct * 16 + mrow;
          xt[(size_t)nn * 64 + n] = acc2[ct][r] + b2s[n];
        }
      }
    }
  }
}

// ===========================================================================
// Kernel 2: fused edge pipeline. Feature order is permuted to
//   feat = [edge_attr(0..15) | SH9(16..24) | pad(25..31)]
// with edge_W1 rows permuted to match, so the attr segment is 16B-aligned.
// Persistent blocks, 16 edges per wave per tile iteration.
// ===========================================================================
__global__ __launch_bounds__(256) void edge_kernel(
    const float* __restrict__ pos, const float* __restrict__ eattr,
    const int* __restrict__ row, const int* __restrict__ col,
    const float* __restrict__ W1, const float* __restrict__ B1,
    const float* __restrict__ W2, const float* __restrict__ B2,
    const float* __restrict__ xt, float* __restrict__ s,
    float* __restrict__ cnt, int Ecnt) {
  __shared__ __align__(16) h16 Wt1[64 * 40];   // [n][k], k padded 25->32, stride 40
  __shared__ __align__(16) h16 Wt2[64 * 72];   // [n][k], k=64, stride 72
  __shared__ float b1s[64], b2s[64];
  __shared__ __align__(16) h16 At[8][16 * 72];
  __shared__ int ridx[8][16], cidx[8][16];

  const int tid = threadIdx.x, w = tid >> 5, lane = tid & 31;
  for (int i = tid; i < 64 * 32; i += 256) {
    int n = i >> 5, k = i & 31;
    // permuted rows: k<16 -> attr row (9+k); 16<=k<25 -> SH row (k-16); pad 0
    float v = 0.f;
    if (k < 16)       v = W1[(9 + k) * 64 + n];
    else if (k < 25)  v = W1[(k - 16) * 64 + n];
    Wt1[n * 40 + k] = (h16)v;
  }
  for (int i = tid; i < 64 * 64; i += 256) {
    int n = i >> 6, k = i & 63;
    Wt2[n * 72 + k] = (h16)W2[k * 64 + n];
  }
  if (tid < 64) { b1s[tid] = B1[tid]; b2s[tid] = B2[tid]; }
  __syncthreads();  // weights ready; everything below is wave-private

  const int mrow = lane & 15;
  const int hi8  = (lane & 16) >> 1;
  const int numTiles = (Ecnt + 15) >> 4;
  const int tileStep = gridDim.x * 8;

  for (int t = blockIdx.x * 8 + w; t < numTiles; t += tileStep) {
    const int e0 = t * 16;
    // prefetch next tile's edge_attr rows (global_prefetch_b8)
    {
      int tn = t + tileStep;
      if (tn * 16 + mrow < Ecnt)
        __builtin_prefetch(eattr + (size_t)(tn * 16 + mrow) * 16, 0, 1);
    }
    if (lane < 16) {  // SH block at halves [16..32)
      int m = lane, e = e0 + m;
      h16* a = &At[w][m * 72];
      h16 tmp[16];
      if (e < Ecnt) {
        int r = row[e], c = col[e];
        ridx[w][m] = r; cidx[w][m] = c;
        float rx = pos[r * 3 + 0] - pos[c * 3 + 0];
        float ry = pos[r * 3 + 1] - pos[c * 3 + 1];
        float rz = pos[r * 3 + 2] - pos[c * 3 + 2];
        float inv1 = __builtin_amdgcn_rsqf(rx * rx + ry * ry + rz * rz + 1e-12f);
        float dx = rx * inv1, dy = ry * inv1, dz = rz * inv1;
        float inv2 = __builtin_amdgcn_rcpf(
            sqrtf(dx * dx + dy * dy + dz * dz) + 1e-10f);
        dx *= inv2; dy *= inv2; dz *= inv2;
        tmp[0] = (h16)0.28209479177387814f;
        tmp[1] = (h16)(0.4886025119029199f * dy);
        tmp[2] = (h16)(0.4886025119029199f * dz);
        tmp[3] = (h16)(0.4886025119029199f * dx);
        tmp[4] = (h16)(1.0925484305920792f * dx * dy);
        tmp[5] = (h16)(1.0925484305920792f * dy * dz);
        tmp[6] = (h16)(0.31539156525252005f * (3.f * dz * dz - 1.f));
        tmp[7] = (h16)(1.0925484305920792f * dx * dz);
        tmp[8] = (h16)(0.5462742152960396f * (dx * dx - dy * dy));
      } else {
        ridx[w][m] = -1; cidx[w][m] = -1;
#pragma unroll
        for (int j = 0; j < 9; ++j) tmp[j] = (h16)0.f;
      }
#pragma unroll
      for (int j = 9; j < 16; ++j) tmp[j] = (h16)0.f;
      ((v8h*)(a + 16))[0] = ((v8h*)tmp)[0];
      ((v8h*)(a + 16))[1] = ((v8h*)tmp)[1];
    } else {  // edge_attr block at halves [0..16), vectorized
      int m = lane - 16, e = e0 + m;
      v8h* dst = (v8h*)&At[w][m * 72];
      if (e < Ecnt) {
        const float4* p = (const float4*)(eattr + (size_t)e * 16);
#pragma unroll
        for (int jj = 0; jj < 2; ++jj) {
          float4 a0 = p[2 * jj], a1 = p[2 * jj + 1];
          v8h hh;
          hh[0] = (h16)a0.x; hh[1] = (h16)a0.y; hh[2] = (h16)a0.z; hh[3] = (h16)a0.w;
          hh[4] = (h16)a1.x; hh[5] = (h16)a1.y; hh[6] = (h16)a1.z; hh[7] = (h16)a1.w;
          dst[jj] = hh;
        }
      } else {
        v8h z = {};
        dst[0] = z; dst[1] = z;
      }
    }

    v8f acc[4] = {v8f{}, v8f{}, v8f{}, v8f{}};
    {
      v16h af = frag_ld(&At[w][mrow * 72], 0, lane);  // K=32: one wmma per ct
#pragma unroll
      for (int ct = 0; ct < 4; ++ct) {
        v16h bf = frag_ld(&Wt1[(ct * 16 + mrow) * 40], 0, lane);
        acc[ct] = wmma16(af, bf, acc[ct]);
      }
    }
#pragma unroll
    for (int ct = 0; ct < 4; ++ct) {
      int n = ct * 16 + mrow;
      float bias = b1s[n];
#pragma unroll
      for (int r = 0; r < 8; ++r) {
        int M = r + hi8;
        At[w][M * 72 + n] = (h16)silu(acc[ct][r] + bias);
      }
    }
    v8f acc2[4] = {v8f{}, v8f{}, v8f{}, v8f{}};
#pragma unroll
    for (int ks = 0; ks < 2; ++ks) {
      v16h af = frag_ld(&At[w][mrow * 72], ks * 32, lane);
#pragma unroll
      for (int ct = 0; ct < 4; ++ct) {
        v16h bf = frag_ld(&Wt2[(ct * 16 + mrow) * 72], ks * 32, lane);
        acc2[ct] = wmma16(af, bf, acc2[ct]);
      }
    }
    // msg = x_t[row] * e_w ; scatter-add into s[col]
#pragma unroll
    for (int r = 0; r < 8; ++r) {
      int M = r + hi8;
      int rn = ridx[w][M], cn = cidx[w][M];
      if (rn >= 0) {
#pragma unroll
        for (int ct = 0; ct < 4; ++ct) {
          int n = ct * 16 + mrow;
          float ew  = acc2[ct][r] + b2s[n];
          float msg = xt[(size_t)rn * 64 + n] * ew;
          atomicAdd(&s[(size_t)cn * 64 + n], msg);
        }
      }
    }
    if (lane < 16) {
      int cn = cidx[w][lane];
      if (cn >= 0) atomicAdd(&cnt[cn], 1.0f);
    }
  }
}

// ===========================================================================
// Kernel 3: out MLP on h = [agg | x] (K=128). Persistent blocks; BN stats
// accumulated in LDS across all tiles, flushed once per block.
// ===========================================================================
__global__ __launch_bounds__(256) void out_mlp_kernel(
    const float* __restrict__ s, const float* __restrict__ cnt,
    const float* __restrict__ x, const float* __restrict__ W1,
    const float* __restrict__ B1, const float* __restrict__ W2,
    const float* __restrict__ B2, float* __restrict__ out,
    float* __restrict__ gsum, float* __restrict__ gsq, int Nn) {
  __shared__ __align__(16) h16 Wt1[64 * 136];  // [n][k], k=0..127, stride 136
  __shared__ __align__(16) h16 Wt2[64 * 72];
  __shared__ float b1s[64], b2s[64];
  __shared__ __align__(16) h16 At[8][16 * 136];
  __shared__ float bsum[64], bsq[64];

  const int tid = threadIdx.x, w = tid >> 5, lane = tid & 31;
  for (int i = tid; i < 64 * 128; i += 256) {
    int n = i >> 7, k = i & 127;
    Wt1[n * 136 + k] = (h16)W1[k * 64 + n];
  }
  for (int i = tid; i < 64 * 64; i += 256) {
    int n = i >> 6, k = i & 63;
    Wt2[n * 72 + k] = (h16)W2[k * 64 + n];
  }
  if (tid < 64) { b1s[tid] = B1[tid]; b2s[tid] = B2[tid]; bsum[tid] = 0.f; bsq[tid] = 0.f; }
  __syncthreads();  // weights + zeroed stats

  const int mrow = lane & 15;
  const int half = lane >> 4;
  const int hi8  = half << 3;
  const int numTiles = (Nn + 15) >> 4;
  const int tileStep = gridDim.x * 8;

  for (int t = blockIdx.x * 8 + w; t < numTiles; t += tileStep) {
    const int n0 = t * 16;
    // --- vectorized A fill: half 0 = agg (s/cnt), half 1 = x ---
    {
      int nn = n0 + mrow;
      v8h* dst = (v8h*)&At[w][mrow * 136 + half * 64];
      if (nn < Nn) {
        float rc = 1.f;
        const float4* p;
        if (half == 0) {
          rc = __builtin_amdgcn_rcpf(fmaxf(cnt[nn], 1.f));
          p = (const float4*)(s + (size_t)nn * 64);
        } else {
          p = (const float4*)(x + (size_t)nn * 64);
        }
#pragma unroll
        for (int jj = 0; jj < 8; ++jj) {
          float4 a0 = p[2 * jj], a1 = p[2 * jj + 1];
          v8h hh;
          hh[0] = (h16)(a0.x * rc); hh[1] = (h16)(a0.y * rc);
          hh[2] = (h16)(a0.z * rc); hh[3] = (h16)(a0.w * rc);
          hh[4] = (h16)(a1.x * rc); hh[5] = (h16)(a1.y * rc);
          hh[6] = (h16)(a1.z * rc); hh[7] = (h16)(a1.w * rc);
          dst[jj] = hh;
        }
      } else {
        v8h z = {};
#pragma unroll
        for (int jj = 0; jj < 8; ++jj) dst[jj] = z;
      }
    }
    v8f acc[4] = {v8f{}, v8f{}, v8f{}, v8f{}};
#pragma unroll
    for (int ks = 0; ks < 4; ++ks) {
      v16h af = frag_ld(&At[w][mrow * 136], ks * 32, lane);
#pragma unroll
      for (int ct = 0; ct < 4; ++ct) {
        v16h bf = frag_ld(&Wt1[(ct * 16 + mrow) * 136], ks * 32, lane);
        acc[ct] = wmma16(af, bf, acc[ct]);
      }
    }
#pragma unroll
    for (int ct = 0; ct < 4; ++ct) {
      int n = ct * 16 + mrow;
      float bias = b1s[n];
#pragma unroll
      for (int r = 0; r < 8; ++r) {
        int M = r + hi8;
        At[w][M * 136 + n] = (h16)silu(acc[ct][r] + bias);
      }
    }
    v8f acc2[4] = {v8f{}, v8f{}, v8f{}, v8f{}};
#pragma unroll
    for (int ks = 0; ks < 2; ++ks) {
      v16h af = frag_ld(&At[w][mrow * 136], ks * 32, lane);
#pragma unroll
      for (int ct = 0; ct < 4; ++ct) {
        v16h bf = frag_ld(&Wt2[(ct * 16 + mrow) * 72], ks * 32, lane);
        acc2[ct] = wmma16(af, bf, acc2[ct]);
      }
    }
#pragma unroll
    for (int r = 0; r < 8; ++r) {
      int M = r + hi8, nn = n0 + M;
      if (nn < Nn) {
#pragma unroll
        for (int ct = 0; ct < 4; ++ct) {
          int n = ct * 16 + mrow;
          float v = acc2[ct][r] + b2s[n];
          out[(size_t)nn * 64 + n] = v;
          atomicAdd(&bsum[n], v);
          atomicAdd(&bsq[n], v * v);
        }
      }
    }
  }
  __syncthreads();
  if (tid < 64) {
    atomicAdd(&gsum[tid], bsum[tid]);
    atomicAdd(&gsq[tid],  bsq[tid]);
  }
}

// ===========================================================================
// Kernel 4: in-place batchnorm of d_out using global sums. Grid stride is a
// multiple of 64, so channel stats are loop-invariant per thread.
// ===========================================================================
__global__ void bn_kernel(float* __restrict__ out, const float* __restrict__ gsum,
                          const float* __restrict__ gsq,
                          const float* __restrict__ gamma,
                          const float* __restrict__ beta, size_t total,
                          float invN) {
  size_t i0 = (size_t)blockIdx.x * blockDim.x + threadIdx.x;
  size_t stride = (size_t)gridDim.x * blockDim.x;  // multiple of 64
  int c = (int)(i0 & 63);
  float mean = gsum[c] * invN;
  float var  = gsq[c] * invN - mean * mean;
  float scl  = __builtin_amdgcn_rsqf(var + 1e-5f) * gamma[c];
  float sh   = beta[c] - mean * scl;
  for (size_t i = i0; i < total; i += stride) {
    out[i] = out[i] * scl + sh;
  }
}

// ===========================================================================
extern "C" void kernel_launch(void* const* d_in, const int* in_sizes, int n_in,
                              void* d_out, int out_size, void* d_ws,
                              size_t ws_size, hipStream_t stream) {
  (void)n_in; (void)out_size; (void)ws_size;
  const float* x       = (const float*)d_in[0];
  const float* pos     = (const float*)d_in[1];
  const float* eattr   = (const float*)d_in[2];
  const int*   row     = (const int*)d_in[3];
  const int*   col     = (const int*)d_in[4];
  const float* nW1     = (const float*)d_in[5];
  const float* nb1     = (const float*)d_in[6];
  const float* nW2     = (const float*)d_in[7];
  const float* nb2     = (const float*)d_in[8];
  const float* eW1     = (const float*)d_in[9];
  const float* eb1     = (const float*)d_in[10];
  const float* eW2     = (const float*)d_in[11];
  const float* eb2     = (const float*)d_in[12];
  const float* oW1     = (const float*)d_in[13];
  const float* ob1     = (const float*)d_in[14];
  const float* oW2     = (const float*)d_in[15];
  const float* ob2     = (const float*)d_in[16];
  const float* bng     = (const float*)d_in[17];
  const float* bnb     = (const float*)d_in[18];

  const int N = in_sizes[0] / 64;
  const int E = in_sizes[3];

  char* ws = (char*)d_ws;
  size_t off = 0;
  float* xt  = (float*)(ws + off); off += (size_t)N * 64 * sizeof(float);
  float* s   = (float*)(ws + off); off += (size_t)N * 64 * sizeof(float);
  float* cnt = (float*)(ws + off); off += (size_t)N * sizeof(float);
  float* gsum = (float*)(ws + off); off += 64 * sizeof(float);
  float* gsq  = (float*)(ws + off); off += 64 * sizeof(float);

  hipMemsetAsync(s, 0, (size_t)N * 64 * sizeof(float), stream);
  hipMemsetAsync(cnt, 0, (size_t)N * sizeof(float), stream);
  hipMemsetAsync(gsum, 0, 64 * sizeof(float), stream);
  hipMemsetAsync(gsq, 0, 64 * sizeof(float), stream);

  // persistent-style grids: amortize the per-block weight stage
  int nodeTiles = (N + 15) / 16;
  int edgeTiles = (E + 15) / 16;
  int nodeBlocks = (nodeTiles + 7) / 8;  if (nodeBlocks > 625)  nodeBlocks = 625;
  int edgeBlocks = (edgeTiles + 7) / 8;  if (edgeBlocks > 1250) edgeBlocks = 1250;

  node_mlp_kernel<<<nodeBlocks, 256, 0, stream>>>(x, nW1, nb1, nW2, nb2, xt, N);
  edge_kernel<<<edgeBlocks, 256, 0, stream>>>(pos, eattr, row, col, eW1, eb1,
                                              eW2, eb2, xt, s, cnt, E);
  out_mlp_kernel<<<nodeBlocks, 256, 0, stream>>>(s, cnt, x, oW1, ob1, oW2, ob2,
                                                 (float*)d_out, gsum, gsq, N);
  size_t total = (size_t)N * 64;
  bn_kernel<<<4096, 256, 0, stream>>>((float*)d_out, gsum, gsq, bng, bnb, total,
                                      1.0f / (float)N);
}
